// SelfAttention_47270410060142
// MI455X (gfx1250) — matrix operand order repaired and checked by reading, not compiled, not run on hardware
//
#include <hip/hip_runtime.h>
#include <stdint.h>

typedef __attribute__((ext_vector_type(16))) __bf16 v16bf;
typedef __attribute__((ext_vector_type(8)))  float  v8f;

#define SEQ    2048
#define DMODEL 2048
#define NQH    16
#define NKVH   4
#define HD     128
#define KVD    (NKVH * HD)

__device__ __forceinline__ unsigned short f2bf(float f) {
  union { float f; uint32_t u; } v; v.f = f;
  uint32_t r = v.u + 0x7FFFu + ((v.u >> 16) & 1u);   // round-to-nearest-even
  return (unsigned short)(r >> 16);
}

__device__ __forceinline__ v8f wmma_bf16(v16bf a, v16bf b, v8f c) {
  // D = A(16x32 bf16) x B(32x16 bf16) + C(16x16 f32)
  return __builtin_amdgcn_wmma_f32_16x16x32_bf16(false, a, false, b, (short)0, c,
                                                 false, false);
}

union Frag { v16bf v; uint32_t u[8]; };

// ---------------------------------------------------------------- convert
__global__ void cvt_f32_bf16(const float* __restrict__ in,
                             unsigned short* __restrict__ out, int n) {
  for (int i = blockIdx.x * blockDim.x + threadIdx.x; i < n;
       i += gridDim.x * blockDim.x)
    out[i] = f2bf(in[i]);
}

// ---------------------------------------------------------------- GEMM
// C[MxN] f32 = A[MxK] bf16 (row-major) @ B[KxN] bf16 (row-major)
// block: 256 thr = 8 waves, tile 128x128, wave tile 32x64, K-step 32
#define BM 128
#define BN 128
#define BK 32

__global__ __launch_bounds__(256) void gemm_bf16_f32(
    const unsigned short* __restrict__ A, const unsigned short* __restrict__ B,
    float* __restrict__ C, int M, int N, int K) {
  __shared__ __align__(16) unsigned short sA[BM * BK];   // [row][k]
  __shared__ __align__(16) unsigned short sBt[BN * BK];  // [n][k] (transposed)
  const int tid = threadIdx.x;
  const int wave = tid >> 5, lane = tid & 31;
  const int lo = lane & 15, hi = lane >> 4;
  const int m0 = blockIdx.x * BM, n0 = blockIdx.y * BN;
  const int wm = (wave >> 1) * 32, wn = (wave & 1) * 64;
  v8f acc[2][4] = {};

  for (int k0 = 0; k0 < K; k0 += BK) {
#pragma unroll
    for (int r = 0; r < 2; ++r) {            // A tile: 128x32, 2 x b128/thread
      int idx = tid + r * 256;
      int row = idx >> 2, seg = idx & 3;
      *(uint4*)(&sA[row * BK + seg * 8]) =
          *(const uint4*)(&A[(size_t)(m0 + row) * K + k0 + seg * 8]);
    }
#pragma unroll
    for (int r = 0; r < 2; ++r) {            // B tile: 32x128 -> sBt[n][k]
      int idx = tid + r * 256;
      int krow = idx >> 4, seg = idx & 15;
      union { uint4 v; unsigned short s[8]; } d;
      d.v = *(const uint4*)(&B[(size_t)(k0 + krow) * N + n0 + seg * 8]);
#pragma unroll
      for (int j = 0; j < 8; ++j) sBt[(seg * 8 + j) * BK + krow] = d.s[j];
    }
    if (k0 + BK < K) {                       // prefetch next tiles into L2/L0
      __builtin_prefetch(&A[(size_t)(m0 + (tid >> 2)) * K + k0 + BK + (tid & 3) * 8], 0, 0);
      __builtin_prefetch(&B[(size_t)(k0 + BK + (tid >> 4)) * N + n0 + (tid & 15) * 8], 0, 0);
    }
    __syncthreads();

    Frag a[2], b[4];
#pragma unroll
    for (int mi = 0; mi < 2; ++mi) {         // A frag: lane lo = row, K pairs
      int row = wm + mi * 16 + lo;
#pragma unroll
      for (int v = 0; v < 8; ++v) {
        int kk = hi * 16 + 2 * v;
        a[mi].u[v] = *(const uint32_t*)(&sA[row * BK + kk]);
      }
    }
#pragma unroll
    for (int ni = 0; ni < 4; ++ni) {         // B frag: lane lo = col, K pairs
      int nn = wn + ni * 16 + lo;
#pragma unroll
      for (int v = 0; v < 8; ++v)
        b[ni].u[v] = *(const uint32_t*)(&sBt[nn * BK + hi * 16 + 2 * v]);
    }
#pragma unroll
    for (int mi = 0; mi < 2; ++mi)
#pragma unroll
      for (int ni = 0; ni < 4; ++ni)
        acc[mi][ni] = wmma_bf16(a[mi].v, b[ni].v, acc[mi][ni]);
    __syncthreads();
  }

#pragma unroll
  for (int mi = 0; mi < 2; ++mi)
#pragma unroll
    for (int ni = 0; ni < 4; ++ni)
#pragma unroll
      for (int v = 0; v < 8; ++v) {          // C layout: M=v+8*hi, N=lo
        int row = m0 + wm + mi * 16 + v + 8 * hi;
        int col = n0 + wn + ni * 16 + lo;
        C[(size_t)row * N + col] = acc[mi][ni][v];
      }
}

// ---------------------------------------------------------------- RoPE + relayout
// in: [SEQ][nh*HD] f32   out: [nh][SEQ][HD] bf16 (rotary applied)
__global__ void rope_to_heads(const float* __restrict__ in,
                              const float* __restrict__ cs,
                              const float* __restrict__ sn,
                              unsigned short* __restrict__ out, int nh) {
  int total = SEQ * nh * (HD / 2);
  for (int idx = blockIdx.x * blockDim.x + threadIdx.x; idx < total;
       idx += gridDim.x * blockDim.x) {
    int i = idx & 63;
    int t = idx >> 6;
    int h = t % nh;
    int s = t / nh;
    size_t ib = (size_t)s * nh * HD + (size_t)h * HD + 2 * i;
    float xr = in[ib], xi = in[ib + 1];
    float c = cs[s * 64 + i], ss = sn[s * 64 + i];
    size_t ob = ((size_t)h * SEQ + s) * HD + 2 * i;
    out[ob]     = f2bf(xr * c - xi * ss);
    out[ob + 1] = f2bf(xr * ss + xi * c);
  }
}

// in: [SEQ][NKVH*HD] f32  out: [NKVH][SEQ][HD] bf16
__global__ void v_to_heads(const float* __restrict__ in,
                           unsigned short* __restrict__ out) {
  int total = SEQ * NKVH * HD;
  for (int idx = blockIdx.x * blockDim.x + threadIdx.x; idx < total;
       idx += gridDim.x * blockDim.x) {
    int d = idx % HD;
    int t = idx / HD;
    int h = t % NKVH;
    int s = t / NKVH;
    out[((size_t)h * SEQ + s) * HD + d] = f2bf(in[idx]);
  }
}

// ---------------------------------------------------------------- flash attention
// grid (SEQ/64, NQH), 128 thr = 4 waves; wave owns 16 q rows; kv streamed 32/iter
__global__ __launch_bounds__(128) void flash_attn_kernel(
    const unsigned short* __restrict__ Q,   // [NQH][SEQ][HD]
    const unsigned short* __restrict__ Kc,  // [NKVH][SEQ][HD]
    const unsigned short* __restrict__ Vc,  // [NKVH][SEQ][HD]
    unsigned short* __restrict__ O) {       // [SEQ][DMODEL] bf16
  __shared__ __align__(16) unsigned short sK[32 * HD];    // [kv][d]
  __shared__ __align__(16) unsigned short sVt[HD * 32];   // [d][kv]
  __shared__ __align__(16) unsigned short sP[4][16 * 32]; // per-wave P scratch
  const int tid = threadIdx.x;
  const int wave = tid >> 5, lane = tid & 31;
  const int lo = lane & 15, hi = lane >> 4;
  const int h = blockIdx.y;
  const int q0 = blockIdx.x * 64;
  const int kvh = h >> 2;                   // N_REP = 4
  const int qrow = q0 + wave * 16;

  Frag qf[4];                               // Q A-fragments over d (4 x K=32)
  {
    const unsigned short* qp = Q + ((size_t)h * SEQ + qrow + lo) * HD;
#pragma unroll
    for (int f = 0; f < 4; ++f)
#pragma unroll
      for (int v = 0; v < 8; ++v) {
        int kk = f * 32 + hi * 16 + 2 * v;
        qf[f].u[v] = *(const uint32_t*)(qp + kk);
      }
  }

  v8f o[8] = {};                            // 8 d-tiles of 16x16 f32
  float m[8], l[8];
#pragma unroll
  for (int v = 0; v < 8; ++v) { m[v] = -3.0e38f; l[v] = 0.f; }
  const float scale = 0.08838834764831845f; // 1/sqrt(128)

  const unsigned short* kbase = Kc + (size_t)kvh * SEQ * HD;
  const unsigned short* vbase = Vc + (size_t)kvh * SEQ * HD;

  for (int kv0 = 0; kv0 < q0 + 64; kv0 += 32) {
#pragma unroll
    for (int r = 0; r < 4; ++r) {           // stage K via CDNA5 async copy,
      int idx = tid + r * 128;              // V via VGPRs with transpose
      int row = idx >> 4, seg = idx & 15;
      unsigned lds_off = (unsigned)(uintptr_t)(&sK[row * HD + seg * 8]);
      const unsigned short* gaddr = &kbase[(size_t)(kv0 + row) * HD + seg * 8];
      asm volatile("global_load_async_to_lds_b128 %0, %1, off"
                   :: "v"(lds_off), "v"(gaddr) : "memory");
      union { uint4 v; unsigned short s[8]; } d;
      d.v = *(const uint4*)(&vbase[(size_t)(kv0 + row) * HD + seg * 8]);
#pragma unroll
      for (int j = 0; j < 8; ++j) sVt[(seg * 8 + j) * 32 + row] = d.s[j];
    }
    asm volatile("s_wait_asynccnt 0" ::: "memory");
    __syncthreads();

    // S = Q @ K^T : two 16x16 tiles (kv lo/hi halves), contraction over d
    v8f sa = {}, sb = {};
#pragma unroll
    for (int f = 0; f < 4; ++f) {
      Frag b0, b1;
#pragma unroll
      for (int v = 0; v < 8; ++v) {
        int dd = f * 32 + hi * 16 + 2 * v;
        b0.u[v] = *(const uint32_t*)(&sK[lo * HD + dd]);
        b1.u[v] = *(const uint32_t*)(&sK[(16 + lo) * HD + dd]);
      }
      sa = wmma_bf16(qf[f].v, b0.v, sa);
      sb = wmma_bf16(qf[f].v, b1.v, sb);
    }

    // causal mask + online softmax (row M = v + 8*hi, col N = lo)
    float alpha[8];
#pragma unroll
    for (int v = 0; v < 8; ++v) {
      int q = qrow + v + 8 * hi;
      float va = sa[v] * scale;
      float vb = sb[v] * scale;
      if (kv0 + lo > q)      va = -3.0e38f;
      if (kv0 + 16 + lo > q) vb = -3.0e38f;
      float r = fmaxf(va, vb);
#pragma unroll
      for (int x = 1; x < 16; x <<= 1) r = fmaxf(r, __shfl_xor(r, x, 32));
      float mn = fmaxf(m[v], r);
      alpha[v] = __expf(m[v] - mn);
      va = __expf(va - mn);
      vb = __expf(vb - mn);
      float rs = va + vb;
#pragma unroll
      for (int x = 1; x < 16; x <<= 1) rs += __shfl_xor(rs, x, 32);
      l[v] = l[v] * alpha[v] + rs;
      m[v] = mn;
      sa[v] = va; sb[v] = vb;
    }

    // P (C-layout) -> LDS -> A-layout fragment; rescale O accumulators
    unsigned short* myP = sP[wave];
#pragma unroll
    for (int v = 0; v < 8; ++v) {
      int mm = v + 8 * hi;
      myP[mm * 32 + lo]      = f2bf(sa[v]);
      myP[mm * 32 + 16 + lo] = f2bf(sb[v]);
    }
#pragma unroll
    for (int t = 0; t < 8; ++t)
#pragma unroll
      for (int v = 0; v < 8; ++v) o[t][v] *= alpha[v];
    asm volatile("s_wait_dscnt 0" ::: "memory");  // wave-local LDS RAW

    Frag pf;
#pragma unroll
    for (int v = 0; v < 8; ++v) {
      int kk = hi * 16 + 2 * v;
      pf.u[v] = *(const uint32_t*)(&myP[lo * 32 + kk]);
    }
#pragma unroll
    for (int t = 0; t < 8; ++t) {           // O += P @ V, 8 d-tiles
      Frag bv;
#pragma unroll
      for (int v = 0; v < 8; ++v)
        bv.u[v] = *(const uint32_t*)(&sVt[(t * 16 + lo) * 32 + hi * 16 + 2 * v]);
      o[t] = wmma_bf16(pf.v, bv.v, o[t]);
    }
    __syncthreads();
  }

#pragma unroll
  for (int v = 0; v < 8; ++v) l[v] = (l[v] > 0.f) ? 1.f / l[v] : 0.f;
#pragma unroll
  for (int t = 0; t < 8; ++t)
#pragma unroll
    for (int v = 0; v < 8; ++v) {
      int row = qrow + v + 8 * hi;
      int col = h * HD + t * 16 + lo;
      O[(size_t)row * DMODEL + col] = f2bf(o[t][v] * l[v]);
    }
}

// ---------------------------------------------------------------- launch
extern "C" void kernel_launch(void* const* d_in, const int* in_sizes, int n_in,
                              void* d_out, int out_size, void* d_ws, size_t ws_size,
                              hipStream_t stream) {
  const float* x  = (const float*)d_in[0];
  const float* fc = (const float*)d_in[1];
  const float* fs = (const float*)d_in[2];
  const float* wq = (const float*)d_in[3];
  const float* wk = (const float*)d_in[4];
  const float* wv = (const float*)d_in[5];
  const float* wo = (const float*)d_in[6];
  float* out = (float*)d_out;

  char* ws = (char*)d_ws;
  size_t off = 0;
  auto take = [&](size_t bytes) -> void* {
    void* p = ws + off;
    off += (bytes + 255) & ~(size_t)255;
    return p;
  };
  unsigned short* xb   = (unsigned short*)take((size_t)SEQ * DMODEL * 2);
  unsigned short* wqb  = (unsigned short*)take((size_t)DMODEL * DMODEL * 2);
  unsigned short* wkb  = (unsigned short*)take((size_t)DMODEL * KVD * 2);
  unsigned short* wvb  = (unsigned short*)take((size_t)DMODEL * KVD * 2);
  unsigned short* wob  = (unsigned short*)take((size_t)DMODEL * DMODEL * 2);
  float* qf            = (float*)take((size_t)SEQ * DMODEL * 4);
  float* kf            = (float*)take((size_t)SEQ * KVD * 4);
  float* vf            = (float*)take((size_t)SEQ * KVD * 4);
  unsigned short* qhb  = (unsigned short*)take((size_t)NQH * SEQ * HD * 2);
  unsigned short* khb  = (unsigned short*)take((size_t)NKVH * SEQ * HD * 2);
  unsigned short* vhb  = (unsigned short*)take((size_t)NKVH * SEQ * HD * 2);
  unsigned short* attb = (unsigned short*)take((size_t)SEQ * DMODEL * 2);

  auto cvt = [&](const float* src, unsigned short* dst, int n) {
    int blocks = (n + 255) / 256;
    if (blocks > 4096) blocks = 4096;
    cvt_f32_bf16<<<blocks, 256, 0, stream>>>(src, dst, n);
  };
  cvt(x,  xb,  SEQ * DMODEL);
  cvt(wq, wqb, DMODEL * DMODEL);
  cvt(wk, wkb, DMODEL * KVD);
  cvt(wv, wvb, DMODEL * KVD);
  cvt(wo, wob, DMODEL * DMODEL);

  gemm_bf16_f32<<<dim3(SEQ / BM, DMODEL / BN), 256, 0, stream>>>(xb, wqb, qf, SEQ, DMODEL, DMODEL);
  gemm_bf16_f32<<<dim3(SEQ / BM, KVD / BN),    256, 0, stream>>>(xb, wkb, kf, SEQ, KVD, DMODEL);
  gemm_bf16_f32<<<dim3(SEQ / BM, KVD / BN),    256, 0, stream>>>(xb, wvb, vf, SEQ, KVD, DMODEL);

  {
    int n = SEQ * NQH * 64;
    int blocks = (n + 255) / 256; if (blocks > 4096) blocks = 4096;
    rope_to_heads<<<blocks, 256, 0, stream>>>(qf, fc, fs, qhb, NQH);
  }
  {
    int n = SEQ * NKVH * 64;
    int blocks = (n + 255) / 256; if (blocks > 4096) blocks = 4096;
    rope_to_heads<<<blocks, 256, 0, stream>>>(kf, fc, fs, khb, NKVH);
  }
  {
    int n = SEQ * NKVH * HD;
    int blocks = (n + 255) / 256; if (blocks > 4096) blocks = 4096;
    v_to_heads<<<blocks, 256, 0, stream>>>(vf, vhb);
  }

  flash_attn_kernel<<<dim3(SEQ / 64, NQH), 128, 0, stream>>>(qhb, khb, vhb, attb);

  gemm_bf16_f32<<<dim3(SEQ / BM, DMODEL / BN), 256, 0, stream>>>(attb, wob, out, SEQ, DMODEL, DMODEL);

  (void)in_sizes; (void)n_in; (void)out_size; (void)ws_size;
}
